// EGNNAreaPlusS_45578192945205
// MI455X (gfx1250) — matrix operand order repaired and checked by reading, not compiled, not run on hardware
//
#include <hip/hip_runtime.h>

// ---------------------------------------------------------------------------
// EGNN (area + dihedral) network for MI455X / gfx1250.
// All MLP GEMMs run on v_wmma_f32_16x16x32_bf16 (wave32 WMMA), f32 accum.
// ---------------------------------------------------------------------------

typedef __bf16 bf16_t;
typedef bf16_t bf16x8  __attribute__((ext_vector_type(8)));
typedef bf16_t bf16x16 __attribute__((ext_vector_type(16)));
typedef float  f32x8   __attribute__((ext_vector_type(8)));

static constexpr int Nv = 60000;
static constexpr int Ne = 360000;
static constexpr int Fc = 120000;
static constexpr int PB = 256;     // partial-reduction blocks

// ---------------------------------------------------------------------------
// WMMA helpers (layouts per CDNA5 ISA 7.12.2)
// ---------------------------------------------------------------------------
static __device__ __forceinline__ bf16x16 cat8(bf16x8 a, bf16x8 b) {
  return __builtin_shufflevector(a, b, 0,1,2,3,4,5,6,7,8,9,10,11,12,13,14,15);
}

static __device__ __forceinline__ f32x8 wmma_bf16(bf16x16 a, bf16x16 b, f32x8 c) {
  return __builtin_amdgcn_wmma_f32_16x16x32_bf16(false, a, false, b, (short)0, c,
                                                 false, false);
}

// A fragment: 16x32 bf16 tile from row-major LDS matrix X (stride kp, kp%32==0).
// lane L (0..15): row=rowBase+L, K in {k0..k0+7, k0+16..k0+23}
// lane L (16..31): row=rowBase+L-16, K in {k0+8..k0+15, k0+24..k0+31}
static __device__ __forceinline__ bf16x16
load_a(const bf16_t* X, int kp, int rowBase, int k0, int lane) {
  const int r  = rowBase + (lane & 15);
  const int kb = k0 + ((lane >> 4) << 3);
  const bf16_t* p = X + r * kp + kb;
  bf16x8 lo = *(const bf16x8*)(p);
  bf16x8 hi = *(const bf16x8*)(p + 16);
  return cat8(lo, hi);
}

// B fragment from packed weights: tile (kt,nt) stored lane-major, 16 bf16/lane.
static __device__ __forceinline__ bf16x16
load_b(const bf16_t* Wp, int kt, int nt, int NT, int lane) {
  const bf16_t* p = Wp + (((size_t)(kt * NT + nt) * 32 + lane) << 4);
  return *(const bf16x16*)p;
}

static __device__ __forceinline__ float act_apply(float v, int act) {
  if (act == 1) return v * __builtin_amdgcn_rcpf(1.f + __expf(-v));  // fast silu
  if (act == 2) return fmaxf(v, 0.f);                                // relu
  return v;
}

// Y[0..31][0..NT*16) = act(X @ Wp + bias), X: LDS 32 x kp bf16. Y: LDS bf16.
// Flattened (nt, mt) tile space distributed over waves for full occupancy.
static __device__ __forceinline__ void
gemm_to_lds(const bf16_t* Xs, int kp, const bf16_t* Wp, const float* bias,
            bf16_t* Ys, int op, int NT, int act) {
  const int lane = threadIdx.x & 31;
  const int wv   = threadIdx.x >> 5;
  const int nw   = blockDim.x >> 5;
  const int KT   = kp >> 5;
  __builtin_prefetch(Wp, 0, 0);
  for (int t = wv; t < NT * 2; t += nw) {
    const int nt = t >> 1, mt = t & 1;
    const int   col = nt * 16 + (lane & 15);
    const float bn  = bias[col];
    f32x8 acc;
#pragma unroll
    for (int i = 0; i < 8; ++i) acc[i] = bn;
    for (int kt = 0; kt < KT; ++kt) {
      bf16x16 a = load_a(Xs, kp, mt * 16, kt * 32, lane);
      bf16x16 b = load_b(Wp, kt, nt, NT, lane);
      acc = wmma_bf16(a, b, acc);
    }
    const int mb = (lane >> 4) << 3;   // D: VGPR r -> M = r (+8 upper half)
#pragma unroll
    for (int i = 0; i < 8; ++i) {
      float v = act_apply(acc[i], act);
      Ys[(mt * 16 + mb + i) * op + col] = (bf16_t)v;
    }
  }
}

// out[row0+..][col] = act(X @ Wp + bias), f32 global store with row/col masks.
static __device__ __forceinline__ void
gemm_to_global(const bf16_t* Xs, int kp, const bf16_t* Wp, const float* bias,
               float* out, int ldo, int ncols, int row0, int nrows, int NT,
               int act) {
  const int lane = threadIdx.x & 31;
  const int wv   = threadIdx.x >> 5;
  const int nw   = blockDim.x >> 5;
  const int KT   = kp >> 5;
  __builtin_prefetch(Wp, 0, 0);
  for (int t = wv; t < NT * 2; t += nw) {
    const int nt = t >> 1, mt = t & 1;
    const int   col = nt * 16 + (lane & 15);
    const float bn  = (col < ncols) ? bias[col] : 0.f;
    f32x8 acc;
#pragma unroll
    for (int i = 0; i < 8; ++i) acc[i] = bn;
    for (int kt = 0; kt < KT; ++kt) {
      bf16x16 a = load_a(Xs, kp, mt * 16, kt * 32, lane);
      bf16x16 b = load_b(Wp, kt, nt, NT, lane);
      acc = wmma_bf16(a, b, acc);
    }
    const int mb = (lane >> 4) << 3;
#pragma unroll
    for (int i = 0; i < 8; ++i) {
      float v  = act_apply(acc[i], act);
      int   gr = row0 + mt * 16 + mb + i;
      if (gr < nrows && col < ncols) out[(size_t)gr * ldo + col] = v;
    }
  }
}

// ---------------------------------------------------------------------------
// Weight packing: f32 [K x Nc] row-major -> bf16 B-fragment-major (zero pad).
// ---------------------------------------------------------------------------
__global__ void pack_w_kernel(const float* __restrict__ W, bf16_t* __restrict__ out,
                              int K, int Nc, int KP, int NP) {
  int idx = blockIdx.x * blockDim.x + threadIdx.x;
  int total = KP * NP;
  if (idx >= total) return;
  int e    = idx & 15;
  int lane = (idx >> 4) & 31;
  int t    = idx >> 9;             // kt*NT + nt
  int NT   = NP >> 4;
  int nt   = t % NT;
  int kt   = t / NT;
  int n = nt * 16 + (lane & 15);
  int k = kt * 32 + ((lane >> 4) << 3) + (e & 7) + ((e >> 3) << 4);
  float v = (k < K && n < Nc) ? W[(size_t)k * Nc + n] : 0.f;
  out[idx] = (bf16_t)v;
}

// ---------------------------------------------------------------------------
// Small utility kernels
// ---------------------------------------------------------------------------
__global__ void fill_f32(float* p, int n) {
  int i = blockIdx.x * blockDim.x + threadIdx.x;
  if (i < n) p[i] = 0.f;
}

__global__ void partial_sum3(const float* __restrict__ pos, float* part, int n) {
  __shared__ float sh[3][256];
  float a0 = 0, a1 = 0, a2 = 0;
  for (int i = blockIdx.x * 256 + threadIdx.x; i < n; i += gridDim.x * 256) {
    a0 += pos[3 * i + 0]; a1 += pos[3 * i + 1]; a2 += pos[3 * i + 2];
  }
  int t = threadIdx.x;
  sh[0][t] = a0; sh[1][t] = a1; sh[2][t] = a2;
  __syncthreads();
  for (int s = 128; s > 0; s >>= 1) {
    if (t < s) { sh[0][t] += sh[0][t+s]; sh[1][t] += sh[1][t+s]; sh[2][t] += sh[2][t+s]; }
    __syncthreads();
  }
  if (t == 0) {
    part[blockIdx.x * 4 + 0] = sh[0][0];
    part[blockIdx.x * 4 + 1] = sh[1][0];
    part[blockIdx.x * 4 + 2] = sh[2][0];
  }
}

__global__ void final_mean(const float* part, float* scal, int nb, int n) {
  if (threadIdx.x < 3) {
    float s = 0;
    for (int i = 0; i < nb; ++i) s += part[i * 4 + threadIdx.x];
    scal[threadIdx.x] = s / (float)n;
  }
}

__global__ void partial_maxsq(const float* __restrict__ pos, const float* scal,
                              float* part, int n) {
  __shared__ float sh[256];
  float m0 = scal[0], m1 = scal[1], m2 = scal[2];
  float mx = 0.f;
  for (int i = blockIdx.x * 256 + threadIdx.x; i < n; i += gridDim.x * 256) {
    float dx = pos[3*i] - m0, dy = pos[3*i+1] - m1, dz = pos[3*i+2] - m2;
    mx = fmaxf(mx, dx*dx + dy*dy + dz*dz);
  }
  int t = threadIdx.x;
  sh[t] = mx;
  __syncthreads();
  for (int s = 128; s > 0; s >>= 1) {
    if (t < s) sh[t] = fmaxf(sh[t], sh[t+s]);
    __syncthreads();
  }
  if (t == 0) part[blockIdx.x] = sh[0];
}

__global__ void final_max(const float* part, float* scal, int nb) {
  if (threadIdx.x == 0) {
    float m = 0.f;
    for (int i = 0; i < nb; ++i) m = fmaxf(m, part[i]);
    scal[3] = 1.f / sqrtf(fmaxf(m, 1e-30f));
  }
}

__global__ void normalize_pos(const float* __restrict__ pos, const float* scal,
                              float* coord, int n3) {
  int t = blockIdx.x * blockDim.x + threadIdx.x;
  if (t < n3) coord[t] = (pos[t] - scal[t % 3]) * scal[3];
}

__global__ void face_area_kernel(const float* __restrict__ coord,
                                 const int* __restrict__ face, float* farea, int fc) {
  int f = blockIdx.x * blockDim.x + threadIdx.x;
  if (f >= fc) return;
  int a = face[f], b = face[fc + f], c = face[2 * fc + f];
  float v1x = coord[3*b]   - coord[3*a],   v1y = coord[3*b+1] - coord[3*a+1],
        v1z = coord[3*b+2] - coord[3*a+2];
  float v2x = coord[3*c]   - coord[3*a],   v2y = coord[3*c+1] - coord[3*a+1],
        v2z = coord[3*c+2] - coord[3*a+2];
  float cx = v1y * v2z - v1z * v2y;
  float cy = v1z * v2x - v1x * v2z;
  float cz = v1x * v2y - v1y * v2x;
  farea[f] = 0.5f * sqrtf(cx*cx + cy*cy + cz*cz);
}

__global__ void scatter_ad(const float* __restrict__ farea,
                           const float* __restrict__ di,
                           const int* __restrict__ v2f,
                           float* area_acc, float* cntv, float* di_acc, int m) {
  int t = blockIdx.x * blockDim.x + threadIdx.x;
  if (t >= m) return;
  int v = v2f[2 * t], f = v2f[2 * t + 1];
  atomicAdd(&area_acc[v], farea[f]);
  atomicAdd(&cntv[v], 1.f);
  if (di_acc) {
    atomicAdd(&di_acc[3*v+0], di[3*f+0]);
    atomicAdd(&di_acc[3*v+1], di[3*f+1]);
    atomicAdd(&di_acc[3*v+2], di[3*f+2]);
  }
}

__global__ void div_ad(float* area_v, float* di_v, const float* cntv, int n) {
  int i = blockIdx.x * blockDim.x + threadIdx.x;
  if (i >= n) return;
  float c = fmaxf(cntv[i], 1.f);
  area_v[i] /= c;
  if (di_v) { di_v[3*i] /= c; di_v[3*i+1] /= c; di_v[3*i+2] /= c; }
}

__global__ void feat_kernel(const float* __restrict__ area_v,
                            const float* __restrict__ hks,
                            const float* __restrict__ W, const float* __restrict__ b,
                            float* out, int n) {
  int t = blockIdx.x * blockDim.x + threadIdx.x;
  if (t >= n * 32) return;
  int v = t >> 5, o = t & 31;
  float s = b[o] + area_v[v] * W[o];
  for (int k = 0; k < 9; ++k) s += hks[v * 9 + k] * W[(k + 1) * 32 + o];
  out[t] = s;
}

__global__ void coord_update(float* coord, const float* trans_acc,
                             const float* cnt, int n) {
  int t = blockIdx.x * blockDim.x + threadIdx.x;
  if (t < 3 * n) coord[t] += trans_acc[t] / fmaxf(cnt[t / 3], 1.f);
}

__global__ void logsoftmax_kernel(float* out, int n, int c) {
  int i = blockIdx.x * blockDim.x + threadIdx.x;
  if (i >= n) return;
  float m = -1e30f;
  for (int j = 0; j < c; ++j) m = fmaxf(m, out[i * c + j]);
  float s = 0.f;
  for (int j = 0; j < c; ++j) s += expf(out[i * c + j] - m);
  float l = logf(s);
  for (int j = 0; j < c; ++j) out[i * c + j] = out[i * c + j] - m - l;
}

// ---------------------------------------------------------------------------
// Fused EGCL edge kernel: 32 edges/block.
//   m = silu(e2(silu(e1([h_i,h_j,radial,w]))));  s = c2(silu(c1(m)))
//   scatter: agg[row]+=m ; trans[row]+=diff*s ; cnt[row]+=1
// ---------------------------------------------------------------------------
template<int NIN, int NHID>
__global__ __launch_bounds__(256) void egcl_edge_kernel(
    const float* __restrict__ h, const float* __restrict__ coord,
    const int* __restrict__ ei, const float* __restrict__ ew,
    const bf16_t* __restrict__ W1, const float* __restrict__ b1,
    const bf16_t* __restrict__ W2, const float* __restrict__ b2,
    const bf16_t* __restrict__ Wc1, const float* __restrict__ bc1,
    const float* __restrict__ wc2, const float* __restrict__ bc2,
    float* __restrict__ agg, float* __restrict__ trans_acc,
    float* __restrict__ cnt, int nE) {
  constexpr int KIN = 2 * NIN + 2;
  constexpr int KP1 = (KIN + 31) & ~31;
  __shared__ alignas(16) bf16_t Xs[32 * KP1];
  __shared__ alignas(16) bf16_t Ys[32 * NHID];
  __shared__ alignas(16) bf16_t Ms[32 * NHID];
  __shared__ float diffs[32 * 3];
  __shared__ float rad[32];
  __shared__ float scl[32];
  __shared__ int   rows[32];
  __shared__ int   cols[32];

  const int tid = threadIdx.x;
  const int e0  = blockIdx.x * 32;

  if (tid < 32) {
    int e = e0 + tid;
    if (e < nE) {
      int r = ei[e], c = ei[nE + e];
      rows[tid] = r; cols[tid] = c;
      float dx = coord[3*r]   - coord[3*c];
      float dy = coord[3*r+1] - coord[3*c+1];
      float dz = coord[3*r+2] - coord[3*c+2];
      diffs[3*tid] = dx; diffs[3*tid+1] = dy; diffs[3*tid+2] = dz;
      rad[tid] = dx*dx + dy*dy + dz*dz;
    } else {
      rows[tid] = -1; cols[tid] = 0;
      diffs[3*tid] = diffs[3*tid+1] = diffs[3*tid+2] = 0.f;
      rad[tid] = 0.f;
    }
  }
  __syncthreads();

  for (int t = tid; t < 32 * KP1; t += blockDim.x) {
    int i = t / KP1, k = t - i * KP1;
    float v = 0.f;
    int r = rows[i];
    if (r >= 0) {
      if (k < NIN)               v = h[(size_t)r * NIN + k];
      else if (k < 2 * NIN)      v = h[(size_t)cols[i] * NIN + (k - NIN)];
      else if (k == 2 * NIN)     v = rad[i];
      else if (k == 2 * NIN + 1) v = ew[e0 + i];
    }
    Xs[t] = (bf16_t)v;
  }
  __syncthreads();

  gemm_to_lds(Xs, KP1, W1, b1, Ys, NHID, NHID / 16, 1);
  __syncthreads();
  gemm_to_lds(Ys, NHID, W2, b2, Ms, NHID, NHID / 16, 1);
  __syncthreads();
  gemm_to_lds(Ms, NHID, Wc1, bc1, Ys, NHID, NHID / 16, 1);
  __syncthreads();

  if (tid < 32) {
    float s = bc2[0];
    for (int k = 0; k < NHID; ++k) s += (float)Ys[tid * NHID + k] * wc2[k];
    scl[tid] = s;
  }
  __syncthreads();

  if (tid < 32 && rows[tid] >= 0) {
    int r = rows[tid];
    float s = scl[tid];
    atomicAdd(&trans_acc[3*r+0], diffs[3*tid+0] * s);
    atomicAdd(&trans_acc[3*r+1], diffs[3*tid+1] * s);
    atomicAdd(&trans_acc[3*r+2], diffs[3*tid+2] * s);
    atomicAdd(&cnt[r], 1.f);
  }
  for (int t = tid; t < 32 * NHID; t += blockDim.x) {
    int i = t / NHID, k = t - i * NHID;
    if (rows[i] >= 0)
      atomicAdd(&agg[(size_t)rows[i] * NHID + k], (float)Ms[i * NHID + k]);
  }
}

// ---------------------------------------------------------------------------
// Fused EGCL node kernel: h_out = n2(silu(n1([h, agg, area, di]))).
// ---------------------------------------------------------------------------
template<int NIN, int NHID, int NOUT>
__global__ __launch_bounds__(256) void egcl_node_kernel(
    const float* __restrict__ h, const float* __restrict__ agg,
    const float* __restrict__ area_v, const float* __restrict__ di_v,
    const bf16_t* __restrict__ Wn1, const float* __restrict__ bn1,
    const bf16_t* __restrict__ Wn2, const float* __restrict__ bn2,
    float* __restrict__ hOut, int n) {
  constexpr int KIN = NIN + NHID + 4;
  constexpr int KP  = (KIN + 31) & ~31;
  __shared__ alignas(16) bf16_t Xs[32 * KP];
  __shared__ alignas(16) bf16_t Ys[32 * NHID];
  const int tid = threadIdx.x;
  const int v0  = blockIdx.x * 32;

  for (int t = tid; t < 32 * KP; t += blockDim.x) {
    int i = t / KP, k = t - i * KP;
    int v = v0 + i;
    float val = 0.f;
    if (v < n) {
      if (k < NIN)                   val = h[(size_t)v * NIN + k];
      else if (k < NIN + NHID)       val = agg[(size_t)v * NHID + (k - NIN)];
      else if (k == NIN + NHID)      val = area_v[v];
      else if (k < NIN + NHID + 4)   val = di_v[3 * v + (k - NIN - NHID - 1)];
    }
    Xs[t] = (bf16_t)val;
  }
  __syncthreads();
  gemm_to_lds(Xs, KP, Wn1, bn1, Ys, NHID, NHID / 16, 1);
  __syncthreads();
  gemm_to_global(Ys, NHID, Wn2, bn2, hOut, NOUT, NOUT, v0, n, NOUT / 16, 0);
}

// ---------------------------------------------------------------------------
// Dense layer: out = act(X @ W + b)
// ---------------------------------------------------------------------------
template<int K, int NP, int NRE, int ACT>
__global__ __launch_bounds__(256) void dense_kernel(
    const float* __restrict__ X, const bf16_t* __restrict__ Wp,
    const float* __restrict__ b, float* __restrict__ out, int nrows) {
  __shared__ alignas(16) bf16_t Xs[32 * K];
  const int tid = threadIdx.x;
  const int r0  = blockIdx.x * 32;
  for (int t = tid; t < 32 * K; t += blockDim.x) {
    int i = t / K, k = t - i * K;
    int r = r0 + i;
    Xs[t] = (bf16_t)((r < nrows) ? X[(size_t)r * K + k] : 0.f);
  }
  __syncthreads();
  gemm_to_global(Xs, K, Wp, b, out, NRE, NRE, r0, nrows, NP / 16, ACT);
}

// ---------------------------------------------------------------------------
// Host side
// ---------------------------------------------------------------------------
template<int NIN, int NHID, int NOUT>
static void run_egcl(hipStream_t st, void* const* d_in, int B,
                     const float* hIn, float* hOut, float* coord,
                     float* agg, float* trans_acc, float* cnt,
                     float* area_v, float* acnt, const float* di_v, float* farea,
                     const int* face, const int* edge, const float* ew,
                     const int* v2f,
                     bf16_t* Pe1, bf16_t* Pe2, bf16_t* Pc1, bf16_t* Pn1,
                     bf16_t* Pn2) {
  const int EB = (Ne + 31) / 32;
  const int NB = (Nv + 31) / 32;
  fill_f32<<<(Nv * NHID + 255) / 256, 256, 0, st>>>(agg, Nv * NHID);
  fill_f32<<<(3 * Nv + 255) / 256, 256, 0, st>>>(trans_acc, 3 * Nv);
  fill_f32<<<(Nv + 255) / 256, 256, 0, st>>>(cnt, Nv);
  egcl_edge_kernel<NIN, NHID><<<EB, 256, 0, st>>>(
      hIn, coord, edge, ew,
      Pe1, (const float*)d_in[B + 1], Pe2, (const float*)d_in[B + 3],
      Pc1, (const float*)d_in[B + 5],
      (const float*)d_in[B + 6], (const float*)d_in[B + 7],
      agg, trans_acc, cnt, Ne);
  coord_update<<<(3 * Nv + 255) / 256, 256, 0, st>>>(coord, trans_acc, cnt, Nv);
  fill_f32<<<(Nv + 255) / 256, 256, 0, st>>>(area_v, Nv);
  fill_f32<<<(Nv + 255) / 256, 256, 0, st>>>(acnt, Nv);
  face_area_kernel<<<(Fc + 255) / 256, 256, 0, st>>>(coord, face, farea, Fc);
  scatter_ad<<<(3 * Fc + 255) / 256, 256, 0, st>>>(farea, nullptr, v2f,
                                                   area_v, acnt, nullptr, 3 * Fc);
  div_ad<<<(Nv + 255) / 256, 256, 0, st>>>(area_v, nullptr, acnt, Nv);
  egcl_node_kernel<NIN, NHID, NOUT><<<NB, 256, 0, st>>>(
      hIn, agg, area_v, di_v,
      Pn1, (const float*)d_in[B + 9], Pn2, (const float*)d_in[B + 11], hOut, Nv);
}

extern "C" void kernel_launch(void* const* d_in, const int* in_sizes, int n_in,
                              void* d_out, int out_size, void* d_ws, size_t ws_size,
                              hipStream_t stream) {
  (void)in_sizes; (void)n_in; (void)out_size; (void)ws_size;

  // --- input indices (setup_inputs dict order, leaf arrays depth-first) ---
  const float* pos    = (const float*)d_in[0];
  const float* hks    = (const float*)d_in[1];
  const float* ew     = (const float*)d_in[2];
  const float* di_ang = (const float*)d_in[3];
  const float* featW  = (const float*)d_in[4];
  const float* featB  = (const float*)d_in[5];
  const int C1 = 6, C2 = 18, C3 = 30;   // conv bases: e1w,e1b,e2w,e2b,c1w,c1b,c2w,c2b,n1w,n1b,n2w,n2b
  const float* lin1b  = (const float*)d_in[43];
  const float* lin2b  = (const float*)d_in[45];
  const int*   face   = (const int*)d_in[46];
  const int*   edge   = (const int*)d_in[47];
  const int*   v2f    = (const int*)d_in[48];

  // --- workspace carve-up (floats) ---
  float* f = (float*)d_ws;
  size_t off = 0;
  auto take = [&](size_t n) { float* p = f + off; off += n; return p; };
  float* coord     = take((size_t)3 * Nv);
  float* hA        = take((size_t)256 * Nv);
  float* hB        = take((size_t)256 * Nv);
  float* agg       = take((size_t)128 * Nv);
  float* trans_acc = take((size_t)3 * Nv);
  float* cnt       = take(Nv);
  float* area_v    = take(Nv);
  float* acnt      = take(Nv);
  float* di_v      = take((size_t)3 * Nv);
  float* farea     = take(Fc);
  float* part      = take((size_t)4 * PB);
  float* part2     = take(PB);
  float* scal      = take(8);

  size_t byteOff = ((off * sizeof(float)) + 255) & ~(size_t)255;
  bf16_t* pool = (bf16_t*)((char*)d_ws + byteOff);

  // --- pack all GEMM weights into bf16 B-fragment layout ---
  struct MD { int idx, K, Nc; };
  const MD mats[17] = {
    {C1 + 0,  66,  32}, {C1 + 2,  32,  32}, {C1 + 4,  32,  32}, {C1 + 8,  68,  32}, {C1 + 10, 32,  64},
    {C2 + 0, 130,  64}, {C2 + 2,  64,  64}, {C2 + 4,  64,  64}, {C2 + 8, 132,  64}, {C2 + 10, 64, 128},
    {C3 + 0, 258, 128}, {C3 + 2, 128, 128}, {C3 + 4, 128, 128}, {C3 + 8, 260, 128}, {C3 + 10,128, 256},
    {42, 256, 256}, {44, 256, 30}
  };
  bf16_t* P[17];
  size_t cur = 0;
  for (int i = 0; i < 17; ++i) {
    int KP = (mats[i].K + 31) & ~31;
    int NP = (mats[i].Nc + 15) & ~15;
    P[i] = pool + cur;
    cur += (size_t)KP * NP;
    int tot = KP * NP;
    pack_w_kernel<<<(tot + 255) / 256, 256, 0, stream>>>(
        (const float*)d_in[mats[i].idx], P[i], mats[i].K, mats[i].Nc, KP, NP);
  }

  // --- position normalization: center then divide by max norm ---
  partial_sum3<<<PB, 256, 0, stream>>>(pos, part, Nv);
  final_mean<<<1, 64, 0, stream>>>(part, scal, PB, Nv);
  partial_maxsq<<<PB, 256, 0, stream>>>(pos, scal, part2, Nv);
  final_max<<<1, 64, 0, stream>>>(part2, scal, PB);
  normalize_pos<<<(3 * Nv + 255) / 256, 256, 0, stream>>>(pos, scal, coord, 3 * Nv);

  // --- initial area_v and di_v (scatter-mean over face corners) ---
  fill_f32<<<(Nv + 255) / 256, 256, 0, stream>>>(area_v, Nv);
  fill_f32<<<(Nv + 255) / 256, 256, 0, stream>>>(acnt, Nv);
  fill_f32<<<(3 * Nv + 255) / 256, 256, 0, stream>>>(di_v, 3 * Nv);
  face_area_kernel<<<(Fc + 255) / 256, 256, 0, stream>>>(coord, face, farea, Fc);
  scatter_ad<<<(3 * Fc + 255) / 256, 256, 0, stream>>>(farea, di_ang, v2f,
                                                       area_v, acnt, di_v, 3 * Fc);
  div_ad<<<(Nv + 255) / 256, 256, 0, stream>>>(area_v, di_v, acnt, Nv);

  // --- feat: x = [area, hks] @ Wf + bf  -> hA [N,32] ---
  feat_kernel<<<(Nv * 32 + 255) / 256, 256, 0, stream>>>(area_v, hks, featW, featB,
                                                         hA, Nv);

  // --- three EGCL layers (32->64->128->256) ---
  run_egcl<32, 32, 64>(stream, d_in, C1, hA, hB, coord, agg, trans_acc, cnt,
                       area_v, acnt, di_v, farea, face, edge, ew, v2f,
                       P[0], P[1], P[2], P[3], P[4]);
  run_egcl<64, 64, 128>(stream, d_in, C2, hB, hA, coord, agg, trans_acc, cnt,
                        area_v, acnt, di_v, farea, face, edge, ew, v2f,
                        P[5], P[6], P[7], P[8], P[9]);
  run_egcl<128, 128, 256>(stream, d_in, C3, hA, hB, coord, agg, trans_acc, cnt,
                          area_v, acnt, di_v, farea, face, edge, ew, v2f,
                          P[10], P[11], P[12], P[13], P[14]);

  // --- head: relu(lin1) -> lin2 -> log_softmax ---
  const int NB = (Nv + 31) / 32;
  dense_kernel<256, 256, 256, 2><<<NB, 256, 0, stream>>>(hB, P[15], lin1b, hA, Nv);
  dense_kernel<256, 32, 30, 0><<<NB, 256, 0, stream>>>(hA, P[16], lin2b,
                                                       (float*)d_out, Nv);
  logsoftmax_kernel<<<(Nv + 255) / 256, 256, 0, stream>>>((float*)d_out, Nv, 30);
}
